// VideoSwinTransformerBlock_4629974745491
// MI455X (gfx1250) — matrix operand order repaired and checked by reading, not compiled, not run on hardware
//
#include <hip/hip_runtime.h>
#include <math.h>

typedef __bf16 bf16;
typedef __attribute__((ext_vector_type(8)))  __bf16 bf16x8;
typedef __attribute__((ext_vector_type(16))) __bf16 bf16x16;
typedef __attribute__((ext_vector_type(8)))  float  f32x8;

#define NWIN   512
#define NTOK   98
#define NPAD   128
#define CCH    256
#define HEADS  8
#define HDIM   32
#define HID    1024
#define MTOT   (NWIN * NPAD)        // 65536 padded tokens
#define MREAL  50176                // 16*56*56
#define QSCALE 0.17677669529663689f // 1/sqrt(32)

__device__ __forceinline__ f32x8 wmma_bf16(bf16x16 a, bf16x16 b, f32x8 c) {
    return __builtin_amdgcn_wmma_f32_16x16x32_bf16(false, a, false, b, (short)0, c,
                                                   false, false);
}
__device__ __forceinline__ void zero8(f32x8& v) {
#pragma unroll
    for (int e = 0; e < 8; ++e) v[e] = 0.0f;
}

// ---------------------------------------------------------------------------
// Generic 2x2-tile WMMA GEMM core:  C(32x32) += A[m0..][K] * Bt[n0..][K]^T
// A row-major [M][K] bf16, Bt row-major [N][K] bf16 (i.e. B transposed).
// ---------------------------------------------------------------------------
__device__ __forceinline__ void gemm_core_2x2(const bf16* __restrict__ A,
                                              const bf16* __restrict__ Bt,
                                              int m0, int n0, int K, f32x8 c[2][2]) {
    const int l  = threadIdx.x & 31;
    const int hi = l >> 4;
    const int kA = hi << 4;   // 0 or 16
    const int kB = hi << 4;   // 0 or 16
    const bf16* a0p = A + (size_t)(m0 + (l & 15)) * K;
    const bf16* a1p = a0p + (size_t)16 * K;
    const bf16* b0p = Bt + (size_t)(n0 + (l & 15)) * K;
    const bf16* b1p = b0p + (size_t)16 * K;
    zero8(c[0][0]); zero8(c[0][1]); zero8(c[1][0]); zero8(c[1][1]);
    for (int k = 0; k < K; k += 32) {
        union { bf16x16 v; bf16x8 h[2]; } a0u, a1u;
        a0u.h[0] = *(const bf16x8*)(a0p + k + kA);
        a0u.h[1] = *(const bf16x8*)(a0p + k + kA + 8);
        a1u.h[0] = *(const bf16x8*)(a1p + k + kA);
        a1u.h[1] = *(const bf16x8*)(a1p + k + kA + 8);
        bf16x16 b0 = *(const bf16x16*)(b0p + k + kB);
        bf16x16 b1 = *(const bf16x16*)(b1p + k + kB);
        c[0][0] = wmma_bf16(a0u.v, b0, c[0][0]);
        c[0][1] = wmma_bf16(a0u.v, b1, c[0][1]);
        c[1][0] = wmma_bf16(a1u.v, b0, c[1][0]);
        c[1][1] = wmma_bf16(a1u.v, b1, c[1][1]);
    }
}

// ---------------------------------------------------------------------------
// Weight convert+transpose: src fp32 [K][N] -> dst bf16 [N][K]; optional col scale
// ---------------------------------------------------------------------------
__global__ void wconv_kernel(const float* __restrict__ src, bf16* __restrict__ dst,
                             int K, int N, int scale_cols, float scale) {
    int idx = blockIdx.x * 256 + threadIdx.x;
    if (idx >= K * N) return;
    int n = idx / K, k = idx % K;
    float v = src[(size_t)k * N + n];
    if (n < scale_cols) v *= scale;
    dst[idx] = (bf16)v;
}

// ---------------------------------------------------------------------------
// LN1 + cyclic shift (-1,-3,-3) + window partition (pad 98->128) -> bf16 A1
// one block (256 thr) per padded token
// ---------------------------------------------------------------------------
__global__ void ln1_part_kernel(const float* __restrict__ x, const float* __restrict__ g,
                                const float* __restrict__ b, bf16* __restrict__ A1) {
    __shared__ float s1[256], s2[256];
    const int blk = blockIdx.x;           // win*128 + t
    const int win = blk >> 7, t = blk & 127;
    const int tid = threadIdx.x;
    const size_t dst = (size_t)blk * CCH + tid;
    if (t >= NTOK) { A1[dst] = (bf16)0.0f; return; }
    const int z = t / 49, rem = t % 49, yy = rem / 7, xx = rem % 7;
    const int db = win >> 6, hb = (win >> 3) & 7, wb = win & 7;
    const int dd = db * 2 + z, hh = hb * 7 + yy, ww = wb * 7 + xx;
    const int d = (dd + 1) & 15, h = (hh + 3) % 56, w = (ww + 3) % 56;
    const size_t src = (((size_t)d * 56 + h) * 56 + w) * CCH + tid;
    const float v = x[src];
    s1[tid] = v; s2[tid] = v * v;
    __syncthreads();
    for (int s = 128; s > 0; s >>= 1) {
        if (tid < s) { s1[tid] += s1[tid + s]; s2[tid] += s2[tid + s]; }
        __syncthreads();
    }
    const float mu  = s1[0] * (1.0f / 256.0f);
    const float var = s2[0] * (1.0f / 256.0f) - mu * mu;
    A1[dst] = (bf16)((v - mu) * rsqrtf(var + 1e-5f) * g[tid] + b[tid]);
}

// ---------------------------------------------------------------------------
// QKV GEMM: A1[65536][256] x Wt[768][256]^T ; scatter into q,k (row-major) and
// v transposed [win*8+head][32][128]. q-scale folded into weights+bias.
// ---------------------------------------------------------------------------
__global__ void gemm_qkv_kernel(const bf16* __restrict__ A1, const bf16* __restrict__ Wt,
                                const float* __restrict__ qkv_b,
                                bf16* __restrict__ q, bf16* __restrict__ k,
                                bf16* __restrict__ vt) {
    const int wid = blockIdx.x * 8 + (threadIdx.x >> 5);
    const int N32 = 768 / 32;
    const int tm = wid / N32, tn = wid % N32;
    f32x8 c[2][2];
    gemm_core_2x2(A1, Wt, tm * 32, tn * 32, CCH, c);
    const int l = threadIdx.x & 31, hi = l >> 4;
#pragma unroll
    for (int mi = 0; mi < 2; ++mi)
#pragma unroll
        for (int ni = 0; ni < 2; ++ni) {
            const int n = tn * 32 + ni * 16 + (l & 15);
            const int part = n >> 8, rem = n & 255, head = rem >> 5, col = rem & 31;
            float bb = qkv_b[n];
            if (part == 0) bb *= QSCALE;
#pragma unroll
            for (int r = 0; r < 8; ++r) {
                const int m   = tm * 32 + mi * 16 + r + (hi << 3);
                const int win = m >> 7, tok = m & 127;
                const float v = c[mi][ni][r] + bb;
                const size_t wh = (size_t)(win * HEADS + head);
                if (part == 0)      q[(wh * NPAD + tok) * HDIM + col] = (bf16)v;
                else if (part == 1) k[(wh * NPAD + tok) * HDIM + col] = (bf16)v;
                else                vt[(wh * HDIM + col) * NPAD + tok] = (bf16)v;
            }
        }
}

// ---------------------------------------------------------------------------
// Fused attention per (window, head): S = qk^T + bias + mask, softmax, O = P v
// block = 256 threads (8 waves); LDS: 64KB S (reused in-place for bf16 P)
// ---------------------------------------------------------------------------
__global__ void attention_kernel(const bf16* __restrict__ q, const bf16* __restrict__ k,
                                 const bf16* __restrict__ vt, const float* __restrict__ rpb,
                                 const float* __restrict__ mask, bf16* __restrict__ o) {
    __shared__ float S[NPAD * NPAD];   // 64 KB
    const int win = blockIdx.x >> 3, head = blockIdx.x & 7;
    const int tid = threadIdx.x, wv = tid >> 5, l = tid & 31, hi = l >> 4;
    const size_t wh = (size_t)(win * HEADS + head);
    const bf16* qb = q + wh * NPAD * HDIM;
    const bf16* kb = k + wh * NPAD * HDIM;
    const bf16* vb = vt + wh * HDIM * NPAD;

    // ---- phase 1: wave wv computes S-tile row wv (16 x 128) ----
    {
        const int mA = wv * 16 + (l & 15);
        const int kA = hi << 4;
        union { bf16x16 v; bf16x8 h[2]; } au;
        au.h[0] = *(const bf16x8*)(qb + mA * HDIM + kA);
        au.h[1] = *(const bf16x8*)(qb + mA * HDIM + kA + 8);
        for (int nt = 0; nt < 8; ++nt) {
            bf16x16 bfrag = *(const bf16x16*)(kb + (nt * 16 + (l & 15)) * HDIM + (hi << 4));
            f32x8 c; zero8(c);
            c = wmma_bf16(au.v, bfrag, c);
            const int j = nt * 16 + (l & 15);
#pragma unroll
            for (int r = 0; r < 8; ++r) {
                const int i = wv * 16 + r + (hi << 3);
                float v = c[r];
                if (i < NTOK && j < NTOK) {
                    const int zi = i / 49, yi = (i % 49) / 7, xi = i % 7;
                    const int zj = j / 49, yj = (j % 49) / 7, xj = j % 7;
                    const int idx = (zi - zj + 1) * 169 + (yi - yj + 6) * 13 + (xi - xj + 6);
                    v += rpb[idx * HEADS + head] +
                         mask[(size_t)win * NTOK * NTOK + i * NTOK + j];
                } else {
                    v = -1e9f;
                }
                S[i * NPAD + j] = v;
            }
        }
    }
    __syncthreads();
    // ---- phase 2: softmax per row; convert to bf16 in place ----
    if (tid < NPAD) {
        float* row = S + tid * NPAD;
        float mx = -3e38f;
        for (int j = 0; j < NPAD; ++j) mx = fmaxf(mx, row[j]);
        float sum = 0.0f;
        for (int j = 0; j < NPAD; ++j) { float e = __expf(row[j] - mx); sum += e; row[j] = e; }
        const float inv = 1.0f / sum;
        bf16* prow = (bf16*)row;       // in-place: bf16 j clobbers float j>>1 (already read)
        for (int j = 0; j < NPAD; ++j) prow[j] = (bf16)(row[j] * inv);
    }
    __syncthreads();
    // ---- phase 3: O = P * V ; wave wv does m-tile wv, n-tiles 0..1 ----
    {
        const bf16* P = (const bf16*)S;    // row stride 256 bf16 slots, first 128 valid
        const int mA = wv * 16 + (l & 15);
        const int kA = hi << 4;
        for (int nt = 0; nt < 2; ++nt) {
            f32x8 c; zero8(c);
            for (int kt = 0; kt < 4; ++kt) {
                union { bf16x16 v; bf16x8 h[2]; } au;
                const bf16* pr = P + (size_t)mA * 256 + kt * 32 + kA;
                au.h[0] = *(const bf16x8*)(pr);
                au.h[1] = *(const bf16x8*)(pr + 8);
                bf16x16 bfrag =
                    *(const bf16x16*)(vb + (nt * 16 + (l & 15)) * NPAD + kt * 32 + (hi << 4));
                c = wmma_bf16(au.v, bfrag, c);
            }
#pragma unroll
            for (int r = 0; r < 8; ++r) {
                const int tok = wv * 16 + r + (hi << 3);
                const int col = nt * 16 + (l & 15);
                o[((size_t)win * NPAD + tok) * CCH + head * HDIM + col] = (bf16)c[r];
            }
        }
    }
}

// ---------------------------------------------------------------------------
// Proj GEMM + window-reverse + un-shift + residual -> x2 fp32 [50176][256]
// ---------------------------------------------------------------------------
__global__ void gemm_proj_kernel(const bf16* __restrict__ o, const bf16* __restrict__ Wt,
                                 const float* __restrict__ pb, const float* __restrict__ x,
                                 float* __restrict__ x2) {
    const int wid = blockIdx.x * 8 + (threadIdx.x >> 5);
    const int N32 = CCH / 32;
    const int tm = wid / N32, tn = wid % N32;
    f32x8 c[2][2];
    gemm_core_2x2(o, Wt, tm * 32, tn * 32, CCH, c);
    const int l = threadIdx.x & 31, hi = l >> 4;
#pragma unroll
    for (int mi = 0; mi < 2; ++mi)
#pragma unroll
        for (int ni = 0; ni < 2; ++ni) {
            const int n = tn * 32 + ni * 16 + (l & 15);
            const float bb = pb[n];
#pragma unroll
            for (int r = 0; r < 8; ++r) {
                const int m   = tm * 32 + mi * 16 + r + (hi << 3);
                const int win = m >> 7, tok = m & 127;
                if (tok >= NTOK) continue;
                const int z = tok / 49, rem = tok % 49, yy = rem / 7, xx = rem % 7;
                const int db = win >> 6, hb = (win >> 3) & 7, wb = win & 7;
                const int dd = db * 2 + z, hh = hb * 7 + yy, ww = wb * 7 + xx;
                const int d = (dd + 1) & 15, h = (hh + 3) % 56, w = (ww + 3) % 56;
                const size_t dst = (((size_t)d * 56 + h) * 56 + w) * CCH + n;
                x2[dst] = x[dst] + c[mi][ni][r] + bb;
            }
        }
}

// ---------------------------------------------------------------------------
// LN2: x2 fp32 [50176][256] -> A2 bf16
// ---------------------------------------------------------------------------
__global__ void ln2_kernel(const float* __restrict__ x2, const float* __restrict__ g,
                           const float* __restrict__ b, bf16* __restrict__ A2) {
    __shared__ float s1[256], s2[256];
    const int tid = threadIdx.x;
    const size_t base = (size_t)blockIdx.x * CCH;
    const float v = x2[base + tid];
    s1[tid] = v; s2[tid] = v * v;
    __syncthreads();
    for (int s = 128; s > 0; s >>= 1) {
        if (tid < s) { s1[tid] += s1[tid + s]; s2[tid] += s2[tid + s]; }
        __syncthreads();
    }
    const float mu  = s1[0] * (1.0f / 256.0f);
    const float var = s2[0] * (1.0f / 256.0f) - mu * mu;
    A2[base + tid] = (bf16)((v - mu) * rsqrtf(var + 1e-5f) * g[tid] + b[tid]);
}

// ---------------------------------------------------------------------------
// FC1 GEMM + exact GELU -> hbuf bf16 [50176][1024]
// ---------------------------------------------------------------------------
__global__ void gemm_fc1_kernel(const bf16* __restrict__ A2, const bf16* __restrict__ Wt,
                                const float* __restrict__ fb, bf16* __restrict__ hbuf) {
    const int wid = blockIdx.x * 8 + (threadIdx.x >> 5);
    const int N32 = HID / 32;
    const int tm = wid / N32, tn = wid % N32;
    f32x8 c[2][2];
    gemm_core_2x2(A2, Wt, tm * 32, tn * 32, CCH, c);
    const int l = threadIdx.x & 31, hi = l >> 4;
#pragma unroll
    for (int mi = 0; mi < 2; ++mi)
#pragma unroll
        for (int ni = 0; ni < 2; ++ni) {
            const int n = tn * 32 + ni * 16 + (l & 15);
            const float bb = fb[n];
#pragma unroll
            for (int r = 0; r < 8; ++r) {
                const int m = tm * 32 + mi * 16 + r + (hi << 3);
                float v = c[mi][ni][r] + bb;
                v = 0.5f * v * (1.0f + erff(v * 0.70710678118654752f));
                hbuf[(size_t)m * HID + n] = (bf16)v;
            }
        }
}

// ---------------------------------------------------------------------------
// FC2 GEMM + residual(x2) -> d_out fp32 [50176][256]
// ---------------------------------------------------------------------------
__global__ void gemm_fc2_kernel(const bf16* __restrict__ hbuf, const bf16* __restrict__ Wt,
                                const float* __restrict__ fb, const float* __restrict__ x2,
                                float* __restrict__ out) {
    const int wid = blockIdx.x * 8 + (threadIdx.x >> 5);
    const int N32 = CCH / 32;
    const int tm = wid / N32, tn = wid % N32;
    f32x8 c[2][2];
    gemm_core_2x2(hbuf, Wt, tm * 32, tn * 32, HID, c);
    const int l = threadIdx.x & 31, hi = l >> 4;
#pragma unroll
    for (int mi = 0; mi < 2; ++mi)
#pragma unroll
        for (int ni = 0; ni < 2; ++ni) {
            const int n = tn * 32 + ni * 16 + (l & 15);
            const float bb = fb[n];
#pragma unroll
            for (int r = 0; r < 8; ++r) {
                const int m = tm * 32 + mi * 16 + r + (hi << 3);
                const size_t idx = (size_t)m * CCH + n;
                out[idx] = x2[idx] + c[mi][ni][r] + bb;
            }
        }
}

// ---------------------------------------------------------------------------
extern "C" void kernel_launch(void* const* d_in, const int* in_sizes, int n_in,
                              void* d_out, int out_size, void* d_ws, size_t ws_size,
                              hipStream_t stream) {
    const float* x      = (const float*)d_in[0];
    const float* maskm  = (const float*)d_in[1];
    const float* g1     = (const float*)d_in[2];
    const float* b1     = (const float*)d_in[3];
    const float* qkv_w  = (const float*)d_in[4];
    const float* qkv_b  = (const float*)d_in[5];
    const float* rpb    = (const float*)d_in[6];
    const float* proj_w = (const float*)d_in[7];
    const float* proj_b = (const float*)d_in[8];
    const float* g2     = (const float*)d_in[9];
    const float* b2     = (const float*)d_in[10];
    const float* fc1_w  = (const float*)d_in[11];
    const float* fc1_b  = (const float*)d_in[12];
    const float* fc2_w  = (const float*)d_in[13];
    const float* fc2_b  = (const float*)d_in[14];
    float* out = (float*)d_out;

    char* w = (char*)d_ws;
    size_t off = 0;
    auto take = [&](size_t bytes) {
        char* p = w + off;
        off += (bytes + 255) & ~(size_t)255;
        return (void*)p;
    };
    bf16* wt_qkv  = (bf16*)take((size_t)768 * 256 * 2);
    bf16* wt_proj = (bf16*)take((size_t)256 * 256 * 2);
    bf16* wt_fc1  = (bf16*)take((size_t)1024 * 256 * 2);
    bf16* wt_fc2  = (bf16*)take((size_t)256 * 1024 * 2);
    bf16* A1 = (bf16*)take((size_t)MTOT * CCH * 2);
    bf16* qb = (bf16*)take((size_t)NWIN * HEADS * NPAD * HDIM * 2);
    bf16* kb = (bf16*)take((size_t)NWIN * HEADS * NPAD * HDIM * 2);
    bf16* vt = (bf16*)take((size_t)NWIN * HEADS * HDIM * NPAD * 2);
    bf16* ob = (bf16*)take((size_t)MTOT * CCH * 2);
    float* x2 = (float*)take((size_t)MREAL * CCH * 4);
    bf16* A2 = (bf16*)take((size_t)MREAL * CCH * 2);
    bf16* hb = (bf16*)take((size_t)MREAL * HID * 2);

    // 1) weight transposes / bf16 conversion (q columns pre-scaled)
    wconv_kernel<<<(256 * 768 + 255) / 256, 256, 0, stream>>>(qkv_w, wt_qkv, 256, 768, 256, QSCALE);
    wconv_kernel<<<(256 * 256 + 255) / 256, 256, 0, stream>>>(proj_w, wt_proj, 256, 256, 0, 1.0f);
    wconv_kernel<<<(256 * 1024 + 255) / 256, 256, 0, stream>>>(fc1_w, wt_fc1, 256, 1024, 0, 1.0f);
    wconv_kernel<<<(1024 * 256 + 255) / 256, 256, 0, stream>>>(fc2_w, wt_fc2, 1024, 256, 0, 1.0f);

    // 2) LN1 + shift + window partition (padded)
    ln1_part_kernel<<<MTOT, 256, 0, stream>>>(x, g1, b1, A1);

    // 3) QKV GEMM  (65536x768x256): tiles32 = 2048*24 = 49152 waves -> 6144 blocks
    gemm_qkv_kernel<<<6144, 256, 0, stream>>>(A1, wt_qkv, qkv_b, qb, kb, vt);

    // 4) fused attention: one block per (window, head)
    attention_kernel<<<NWIN * HEADS, 256, 0, stream>>>(qb, kb, vt, rpb, maskm, ob);

    // 5) proj GEMM + reverse + residual (65536x256x256): 2048*8 = 16384 waves -> 2048 blocks
    gemm_proj_kernel<<<2048, 256, 0, stream>>>(ob, wt_proj, proj_b, x, x2);

    // 6) LN2
    ln2_kernel<<<MREAL / 1, 256, 0, stream>>>(x2, g2, b2, A2);

    // 7) FC1 + GELU (50176x1024x256): 1568*32 = 50176 waves -> 6272 blocks
    gemm_fc1_kernel<<<6272, 256, 0, stream>>>(A2, wt_fc1, fc1_b, hb);

    // 8) FC2 + residual (50176x256x1024): 1568*8 = 12544 waves -> 1568 blocks
    gemm_fc2_kernel<<<1568, 256, 0, stream>>>(hb, wt_fc2, fc2_b, x2, out);
}
